// DecoderLayer_25494925869174
// MI455X (gfx1250) — matrix operand order repaired and checked by reading, not compiled, not run on hardware
//
#include <hip/hip_runtime.h>

#define D_MODEL 1024
#define NUM_HEADS 16
#define HEAD_D 64
#define D_FF 4096
#define BATCH 4
#define SEQ 1024
#define BS (BATCH * SEQ)            // 4096 rows
#define LN_EPS 1e-5f
#define QROWS 32                    // q-rows per fused-attention block

typedef __bf16 bf16;
typedef __attribute__((ext_vector_type(8)))  __bf16 v8bf;
typedef __attribute__((ext_vector_type(16))) __bf16 v16bf;
typedef __attribute__((ext_vector_type(8)))  float  v8f;

// ---------------------------------------------------------------------------
// WMMA fragment loaders (CDNA5 ISA 7.12.2 VGPR layouts, wave32)
// A: 16x32 bf16. lane(0-15): M=lane, halves[0..7]=K(8g..8g+7), [8..15]=K(16+8g..)
// B: 32x16 bf16 (from Bt, row-major NxK). lane: N=lane&15, halves[0..7]=K(8g..8g+7), [8..15]=K(16+8g..)
// ---------------------------------------------------------------------------
__device__ __forceinline__ v16bf load_frag_a(const bf16* __restrict__ row, int k0, int g) {
  v8bf lo = *(const v8bf*)(row + k0 + 8 * g);
  v8bf hi = *(const v8bf*)(row + k0 + 16 + 8 * g);
  return __builtin_shufflevector(lo, hi, 0,1,2,3,4,5,6,7,8,9,10,11,12,13,14,15);
}
__device__ __forceinline__ v16bf load_frag_b(const bf16* __restrict__ row, int k0, int g) {
  v8bf lo = *(const v8bf*)(row + k0 + 8 * g);
  v8bf hi = *(const v8bf*)(row + k0 + 16 + 8 * g);
  return __builtin_shufflevector(lo, hi, 0,1,2,3,4,5,6,7,8,9,10,11,12,13,14,15);
}

// ---------------------------------------------------------------------------
// Generic NT GEMM: C[M,N] = scale * A[M,K] * Bt[N,K]^T (+bias) (+ReLU)
// A, Bt bf16 row-major (K contiguous). 256 threads = 8 waves; 64x64 tile.
// Wave w: rows 16*(w&3), cols 32*(w>>2) + {0,16}. Batched via grid.z:
// z -> (zb = z/zdiv, zi = z%zdiv), pointer += zb*s?0 + zi*s?1.
// ---------------------------------------------------------------------------
template <typename OutT, bool RELU>
__global__ __launch_bounds__(256) void gemm_nt(
    const bf16* __restrict__ A, const bf16* __restrict__ Bt,
    const float* __restrict__ bias, OutT* __restrict__ C,
    int K, int lda, int ldb, int ldc, int zdiv,
    long long sA0, long long sA1, long long sB0, long long sB1,
    long long sC0, long long sC1, float scale) {
  const int z  = blockIdx.z;
  const int zb = z / zdiv, zi = z % zdiv;
  A  += zb * sA0 + zi * sA1;
  Bt += zb * sB0 + zi * sB1;
  C  += zb * sC0 + zi * sC1;

  const int lane = threadIdx.x & 31;
  const int wave = threadIdx.x >> 5;
  const int m16  = lane & 15;
  const int g    = lane >> 4;

  const int rowBase = blockIdx.y * 64 + (wave & 3) * 16;
  const int colBase = blockIdx.x * 64 + (wave >> 2) * 32;

  const bf16* arow  = A  + (long long)(rowBase + m16) * lda;
  const bf16* brow0 = Bt + (long long)(colBase + m16) * ldb;
  const bf16* brow1 = brow0 + (long long)16 * ldb;

  v8f acc0 = {};
  v8f acc1 = {};
  for (int k0 = 0; k0 < K; k0 += 32) {
    v16bf af = load_frag_a(arow, k0, g);
    v16bf b0 = load_frag_b(brow0, k0, g);
    v16bf b1 = load_frag_b(brow1, k0, g);
    acc0 = __builtin_amdgcn_wmma_f32_16x16x32_bf16(false, af, false, b0, (short)0, acc0, false, false);
    acc1 = __builtin_amdgcn_wmma_f32_16x16x32_bf16(false, af, false, b1, (short)0, acc1, false, false);
  }

  const int c0 = colBase + m16;
  const int c1 = c0 + 16;
  float bb0 = bias ? bias[c0] : 0.0f;
  float bb1 = bias ? bias[c1] : 0.0f;
#pragma unroll
  for (int r = 0; r < 8; ++r) {
    int mm   = rowBase + 8 * g + r;
    float v0 = acc0[r] * scale + bb0;
    float v1 = acc1[r] * scale + bb1;
    if (RELU) { v0 = fmaxf(v0, 0.0f); v1 = fmaxf(v1, 0.0f); }
    OutT* crow = C + (long long)mm * ldc;
    crow[c0] = (OutT)v0;
    crow[c1] = (OutT)v1;
  }
}

// ---------------------------------------------------------------------------
// Fused attention probabilities: P = softmax(Q K^T / sqrt(d) + mask)
// One block = QROWS q-rows x full Sk=SEQ for one (b,h). Logits tile lives
// entirely in LDS (QROWS*SEQ*4 = 128 KB; MI455X WGP has 320 KB).
// fp32 P -> d_out region (the validated output), bf16 P -> ws for the PV GEMM.
// Q tile (QROWS x 64 bf16) is staged to LDS with the CDNA5 async-to-LDS path.
// grid = (SEQ/QROWS, B*H), block = 256 (8 waves).
// ---------------------------------------------------------------------------
__global__ __launch_bounds__(256) void attn_fused(
    const bf16* __restrict__ Q, const bf16* __restrict__ Kp,
    const float* __restrict__ mask,
    float* __restrict__ probs_f, bf16* __restrict__ probs_b) {
  extern __shared__ char smem_raw[];
  float* Lg = (float*)smem_raw;                                  // [QROWS][SEQ]
  bf16*  Qs = (bf16*)(smem_raw + (size_t)QROWS * SEQ * sizeof(float)); // [QROWS][64]

  const int z = blockIdx.y;                 // b*H + h
  const int b = z / NUM_HEADS, h = z % NUM_HEADS;
  const int qBase = blockIdx.x * QROWS;

  const bf16* Qg = Q  + ((long long)b * SEQ + qBase) * D_MODEL + h * HEAD_D;
  const bf16* Kg = Kp + (long long)b * SEQ * D_MODEL + h * HEAD_D;

  // ---- stage Q tile: QROWS rows x 128 B = 256 x 16 B chunks, 1 per thread ----
  {
    const int t   = threadIdx.x;
    const int row = t >> 3;             // 0..31
    const int col = (t & 7) * 8;        // bf16 elements, 16 B chunks
    const bf16* src = Qg + (long long)row * D_MODEL + col;
    unsigned int ldsoff = (unsigned int)(uintptr_t)(Qs + row * HEAD_D + col);
    long long    gaddr  = (long long)(uintptr_t)src;
    asm volatile("global_load_async_to_lds_b128 %0, %1, off"
                 :: "v"(ldsoff), "v"(gaddr) : "memory");
    asm volatile("s_wait_asynccnt 0x0" ::: "memory");
  }
  __syncthreads();

  const int lane = threadIdx.x & 31;
  const int wave = threadIdx.x >> 5;
  const int m16  = lane & 15;
  const int g    = lane >> 4;

  // ---- compute logits tiles: 2 row-subtiles x 64 col-subtiles, 16 per wave ----
  const int rowBase = (wave & 1) * 16;
  const bf16* arow = Qs + (rowBase + m16) * HEAD_D;
  v16bf a0 = load_frag_a(arow, 0, g);
  v16bf a1 = load_frag_a(arow, 32, g);

  const float* Mbase = mask + ((long long)b * SEQ + qBase) * SEQ;

  for (int j = 0; j < 16; ++j) {
    const int colBase = ((wave >> 1) * 16 + j) * 16;
    const bf16* brow = Kg + (long long)(colBase + m16) * D_MODEL;
    v16bf b0 = load_frag_b(brow, 0, g);
    v16bf b1 = load_frag_b(brow, 32, g);
    v8f acc = {};
    acc = __builtin_amdgcn_wmma_f32_16x16x32_bf16(false, a0, false, b0, (short)0, acc, false, false);
    acc = __builtin_amdgcn_wmma_f32_16x16x32_bf16(false, a1, false, b1, (short)0, acc, false, false);
    const int c = colBase + m16;
#pragma unroll
    for (int r = 0; r < 8; ++r) {
      const int mLoc = rowBase + 8 * g + r;
      Lg[mLoc * SEQ + c] = acc[r] * 0.125f + Mbase[(long long)mLoc * SEQ + c];
    }
  }
  __syncthreads();

  // ---- softmax: 4 rows per wave, 32 elems per lane, shuffle reductions ----
  for (int rr = 0; rr < 4; ++rr) {
    const int r = wave * 4 + rr;
    const float* Lrow = Lg + r * SEQ;
    float vals[32];
    float mx = -INFINITY;
#pragma unroll
    for (int i = 0; i < 32; ++i) {
      vals[i] = Lrow[lane + i * 32];
      mx = fmaxf(mx, vals[i]);
    }
#pragma unroll
    for (int off = 16; off > 0; off >>= 1) mx = fmaxf(mx, __shfl_xor(mx, off, 32));
    float sum = 0.0f;
#pragma unroll
    for (int i = 0; i < 32; ++i) {
      vals[i] = __expf(vals[i] - mx);
      sum += vals[i];
    }
#pragma unroll
    for (int off = 16; off > 0; off >>= 1) sum += __shfl_xor(sum, off, 32);
    const float inv = 1.0f / sum;
    const long long orow = ((long long)z * SEQ + (qBase + r)) * SEQ;
    float* Pf = probs_f + orow;
    bf16*  Pb = probs_b + orow;
#pragma unroll
    for (int i = 0; i < 32; ++i) {
      const float p = vals[i] * inv;
      Pf[lane + i * 32] = p;
      Pb[lane + i * 32] = (bf16)p;
    }
  }
}

// ---------------------------------------------------------------------------
// f32 -> bf16 elementwise cast
// ---------------------------------------------------------------------------
__global__ __launch_bounds__(256) void cast_bf16(const float* __restrict__ in,
                                                 bf16* __restrict__ out, long long n) {
  long long i = (long long)blockIdx.x * blockDim.x + threadIdx.x;
  long long stride = (long long)gridDim.x * blockDim.x;
  for (; i < n; i += stride) out[i] = (bf16)in[i];
}

// ---------------------------------------------------------------------------
// Transpose + cast: W[K,N] f32 -> Wt[N,K] bf16 (32x32 LDS tile, padded)
// ---------------------------------------------------------------------------
__global__ __launch_bounds__(256) void transpose_cast(const float* __restrict__ W,
                                                      bf16* __restrict__ Wt,
                                                      int K, int N) {
  __shared__ float tile[32][33];
  const int tx = threadIdx.x, ty = threadIdx.y;
  const int x = blockIdx.x * 32 + tx;  // N dim
#pragma unroll
  for (int i = 0; i < 4; ++i) {
    int y = blockIdx.y * 32 + ty + i * 8;  // K dim
    tile[ty + i * 8][tx] = W[(long long)y * N + x];
  }
  __syncthreads();
  const int xo = blockIdx.y * 32 + tx;  // K dim
#pragma unroll
  for (int i = 0; i < 4; ++i) {
    int yo = blockIdx.x * 32 + ty + i * 8;  // N dim
    Wt[(long long)yo * K + xo] = (bf16)tile[tx][ty + i * 8];
  }
}

// ---------------------------------------------------------------------------
// Per-head V transpose: V (B,S,H,d) bf16 -> Vt (B,H,d,S) bf16, LDS-tiled.
// grid = (SEQ/32, HEAD_D/32, B*H), block = (32,8)
// ---------------------------------------------------------------------------
__global__ __launch_bounds__(256) void transpose_head_v(const bf16* __restrict__ v,
                                                        bf16* __restrict__ vt) {
  __shared__ bf16 tile[32][33];
  const int z = blockIdx.z;
  const int b = z / NUM_HEADS, h = z % NUM_HEADS;
  const bf16* in = v + (long long)b * SEQ * D_MODEL + h * HEAD_D;
  bf16* out = vt + (long long)z * HEAD_D * SEQ;
  const int tx = threadIdx.x, ty = threadIdx.y;
  const int s0 = blockIdx.x * 32;  // seq tile
  const int d0 = blockIdx.y * 32;  // head-dim tile
#pragma unroll
  for (int i = 0; i < 4; ++i)
    tile[ty + i * 8][tx] = in[(long long)(s0 + ty + i * 8) * D_MODEL + d0 + tx];
  __syncthreads();
#pragma unroll
  for (int i = 0; i < 4; ++i)
    out[(long long)(d0 + ty + i * 8) * SEQ + s0 + tx] = tile[tx][ty + i * 8];
}

// ---------------------------------------------------------------------------
// out = LayerNorm(a + res) * g + beta ; optional bf16 copy. One row per block.
// ---------------------------------------------------------------------------
__global__ __launch_bounds__(256) void add_layernorm(const float* __restrict__ a,
                                                     const float* __restrict__ res,
                                                     const float* __restrict__ gam,
                                                     const float* __restrict__ bet,
                                                     float* __restrict__ out,
                                                     bf16* __restrict__ out_bf) {
  const long long row = blockIdx.x;
  const float* A = a   + row * D_MODEL;
  const float* R = res + row * D_MODEL;
  float* O = out + row * D_MODEL;
  __shared__ float red[256];
  __shared__ float stats[2];
  const int tid = threadIdx.x;

  float v[4];
  float s = 0.0f;
#pragma unroll
  for (int i = 0; i < 4; ++i) {
    int c = tid + i * 256;
    v[i] = A[c] + R[c];
    s += v[i];
  }
  red[tid] = s;
  __syncthreads();
  for (int t = 128; t > 0; t >>= 1) {
    if (tid < t) red[tid] += red[tid + t];
    __syncthreads();
  }
  if (tid == 0) stats[0] = red[0] * (1.0f / D_MODEL);
  __syncthreads();
  const float mu = stats[0];
  float sq = 0.0f;
#pragma unroll
  for (int i = 0; i < 4; ++i) {
    float d = v[i] - mu;
    sq += d * d;
  }
  __syncthreads();  // red[] reuse
  red[tid] = sq;
  __syncthreads();
  for (int t = 128; t > 0; t >>= 1) {
    if (tid < t) red[tid] += red[tid + t];
    __syncthreads();
  }
  if (tid == 0) stats[1] = rsqrtf(red[0] * (1.0f / D_MODEL) + LN_EPS);
  __syncthreads();
  const float rs = stats[1];
#pragma unroll
  for (int i = 0; i < 4; ++i) {
    int c = tid + i * 256;
    float y = (v[i] - mu) * rs * gam[c] + bet[c];
    O[c] = y;
    if (out_bf) out_bf[row * D_MODEL + c] = (bf16)y;
  }
}

// ===========================================================================
// Host side
// ===========================================================================
extern "C" void kernel_launch(void* const* d_in, const int* in_sizes, int n_in,
                              void* d_out, int out_size, void* d_ws, size_t ws_size,
                              hipStream_t stream) {
  (void)in_sizes; (void)n_in; (void)out_size; (void)ws_size;

  const float* x_f    = (const float*)d_in[0];
  const float* enc_f  = (const float*)d_in[1];
  const float* padM   = (const float*)d_in[2];   // cross-attn mask
  const float* seqM   = (const float*)d_in[3];   // self-attn mask
  const float* Ww[10] = {(const float*)d_in[4],  (const float*)d_in[6],
                         (const float*)d_in[8],  (const float*)d_in[10],
                         (const float*)d_in[12], (const float*)d_in[14],
                         (const float*)d_in[16], (const float*)d_in[18],
                         (const float*)d_in[20], (const float*)d_in[22]};
  const float* Wb[10] = {(const float*)d_in[5],  (const float*)d_in[7],
                         (const float*)d_in[9],  (const float*)d_in[11],
                         (const float*)d_in[13], (const float*)d_in[15],
                         (const float*)d_in[17], (const float*)d_in[19],
                         (const float*)d_in[21], (const float*)d_in[23]};
  const float* ln_g[3] = {(const float*)d_in[24], (const float*)d_in[26], (const float*)d_in[28]};
  const float* ln_b[3] = {(const float*)d_in[25], (const float*)d_in[27], (const float*)d_in[29]};

  float* out3  = (float*)d_out;
  float* attn1 = out3  + (long long)BS * D_MODEL;                       // (B,H,S,S)
  float* attn2 = attn1 + (long long)BATCH * NUM_HEADS * SEQ * SEQ;      // (B,H,S,S)

  // ---- workspace layout ----
  char* ws = (char*)d_ws;
  size_t cur = 0;
  auto walloc = [&](size_t bytes) -> char* {
    cur = (cur + 255) & ~(size_t)255;
    char* p = ws + cur;
    cur += bytes;
    return p;
  };
  const size_t ACT_BF = (size_t)BS * D_MODEL * sizeof(bf16);   // 8 MB
  const size_t ACT_F  = (size_t)BS * D_MODEL * sizeof(float);  // 16 MB

  bf16* wt[10];
  for (int i = 0; i < 8; ++i) wt[i] = (bf16*)walloc((size_t)D_MODEL * D_MODEL * sizeof(bf16));
  wt[8] = (bf16*)walloc((size_t)D_MODEL * D_FF * sizeof(bf16));  // ffn1t (D_FF x D_MODEL)
  wt[9] = (bf16*)walloc((size_t)D_FF * D_MODEL * sizeof(bf16));  // ffn2t (D_MODEL x D_FF)

  bf16* x_bf   = (bf16*)walloc(ACT_BF);
  bf16* enc_bf = (bf16*)walloc(ACT_BF);
  bf16* qb     = (bf16*)walloc(ACT_BF);
  bf16* kb     = (bf16*)walloc(ACT_BF);
  bf16* vb     = (bf16*)walloc(ACT_BF);
  bf16* vt     = (bf16*)walloc(ACT_BF);
  bf16* attnb  = (bf16*)walloc(ACT_BF);
  bf16* probs  = (bf16*)walloc((size_t)BATCH * NUM_HEADS * SEQ * SEQ * sizeof(bf16)); // 128 MB
  float* projf = (float*)walloc(ACT_F);
  float* out1  = (float*)walloc(ACT_F);
  bf16*  out1b = (bf16*)walloc(ACT_BF);
  float* out2  = (float*)walloc(ACT_F);
  bf16*  out2b = (bf16*)walloc(ACT_BF);
  bf16*  f1b   = (bf16*)walloc((size_t)BS * D_FF * sizeof(bf16));  // 32 MB

  // ---- GEMM launch helpers ----
  auto gemm_plain = [&](const bf16* A, const bf16* Bt, const float* bias, bf16* C,
                        int M, int N, int K, int lda, int ldb, int ldc, bool relu) {
    dim3 g(N / 64, M / 64, 1), b(256);
    if (relu)
      gemm_nt<bf16, true><<<g, b, 0, stream>>>(A, Bt, bias, C, K, lda, ldb, ldc, 1,
                                               0, 0, 0, 0, 0, 0, 1.0f);
    else
      gemm_nt<bf16, false><<<g, b, 0, stream>>>(A, Bt, bias, C, K, lda, ldb, ldc, 1,
                                                0, 0, 0, 0, 0, 0, 1.0f);
  };
  auto gemm_f32 = [&](const bf16* A, const bf16* Bt, const float* bias, float* C,
                      int M, int N, int K, int lda, int ldb, int ldc) {
    dim3 g(N / 64, M / 64, 1), b(256);
    gemm_nt<float, false><<<g, b, 0, stream>>>(A, Bt, bias, C, K, lda, ldb, ldc, 1,
                                               0, 0, 0, 0, 0, 0, 1.0f);
  };

  // ---- 1) weight transpose+cast (once per matrix) ----
  {
    dim3 b(32, 8);
    for (int i = 0; i < 8; ++i)
      transpose_cast<<<dim3(D_MODEL / 32, D_MODEL / 32), b, 0, stream>>>(Ww[i], wt[i], D_MODEL, D_MODEL);
    transpose_cast<<<dim3(D_FF / 32, D_MODEL / 32), b, 0, stream>>>(Ww[8], wt[8], D_MODEL, D_FF);
    transpose_cast<<<dim3(D_MODEL / 32, D_FF / 32), b, 0, stream>>>(Ww[9], wt[9], D_FF, D_MODEL);
  }
  // ---- 2) activation casts ----
  {
    long long n = (long long)BS * D_MODEL;
    cast_bf16<<<2048, 256, 0, stream>>>(x_f, x_bf, n);
    cast_bf16<<<2048, 256, 0, stream>>>(enc_f, enc_bf, n);
  }

  const long long SD  = (long long)SEQ * D_MODEL;        // per-batch activation stride
  const long long SS  = (long long)SEQ * SEQ;            // per-head probs stride
  const long long HSS = (long long)NUM_HEADS * SS;       // per-batch probs stride
  const long long HDS = (long long)HEAD_D * SEQ;         // per-head Vt stride
  const size_t SMEM_ATTN = (size_t)QROWS * SEQ * sizeof(float) +
                           (size_t)QROWS * HEAD_D * sizeof(bf16);  // 132 KB

  auto run_mha = [&](const bf16* q_src, const bf16* kv_src, int wbase,
                     float* attn_out, const float* mask) {
    // projections (M = B*S rows, batch-agnostic)
    gemm_plain(q_src,  wt[wbase + 0], Wb[wbase + 0], qb, BS, D_MODEL, D_MODEL, D_MODEL, D_MODEL, D_MODEL, false);
    gemm_plain(kv_src, wt[wbase + 1], Wb[wbase + 1], kb, BS, D_MODEL, D_MODEL, D_MODEL, D_MODEL, D_MODEL, false);
    gemm_plain(kv_src, wt[wbase + 2], Wb[wbase + 2], vb, BS, D_MODEL, D_MODEL, D_MODEL, D_MODEL, D_MODEL, false);
    transpose_head_v<<<dim3(SEQ / 32, HEAD_D / 32, BATCH * NUM_HEADS), dim3(32, 8), 0, stream>>>(vb, vt);
    // fused logits + mask + softmax: fp32 -> d_out, bf16 -> ws
    attn_fused<<<dim3(SEQ / QROWS, BATCH * NUM_HEADS), 256, SMEM_ATTN, stream>>>(
        qb, kb, mask, attn_out, probs);
    // attn = P V  -> (B,S,H,d) bf16
    {
      dim3 g(HEAD_D / 64, SEQ / 64, BATCH * NUM_HEADS), b(256);
      gemm_nt<bf16, false><<<g, b, 0, stream>>>(probs, vt, nullptr, attnb,
          SEQ, SEQ, SEQ, D_MODEL, NUM_HEADS,
          HSS, SS, HDS * NUM_HEADS, HDS, SD, (long long)HEAD_D, 1.0f);
    }
    // output projection -> fp32
    gemm_f32(attnb, wt[wbase + 3], Wb[wbase + 3], projf, BS, D_MODEL, D_MODEL, D_MODEL, D_MODEL, D_MODEL);
  };

  // ---- MHA1 (self) + LN1 ----
  run_mha(x_bf, x_bf, 0, attn1, seqM);
  add_layernorm<<<BS, 256, 0, stream>>>(projf, x_f, ln_g[0], ln_b[0], out1, out1b);

  // ---- MHA2 (cross) + LN2 ----
  run_mha(out1b, enc_bf, 4, attn2, padM);
  add_layernorm<<<BS, 256, 0, stream>>>(projf, out1, ln_g[1], ln_b[1], out2, out2b);

  // ---- FFN + LN3 ----
  gemm_plain(out2b, wt[8], Wb[8], f1b, BS, D_FF, D_MODEL, D_MODEL, D_MODEL, D_FF, /*relu=*/true);
  gemm_f32(f1b, wt[9], Wb[9], projf, BS, D_MODEL, D_FF, D_FF, D_FF, D_MODEL);
  add_layernorm<<<BS, 256, 0, stream>>>(projf, out2, ln_g[2], ln_b[2], out3, nullptr);
}
